// DEERForward_58737972740372
// MI455X (gfx1250) — matrix operand order repaired and checked by reading, not compiled, not run on hardware
//
#include <hip/hip_runtime.h>
#include <hip/hip_bf16.h>
#include <stdint.h>

// ---------------------------------------------------------------------------
// Problem constants (from reference: L=12, B=2, T=512, C=768, F=3072)
// ---------------------------------------------------------------------------
#define L_LAYERS 12
#define BT       1024                 // B*T
#define CDIM     768
#define FDIM     3072
#define S_ELEMS  (BT * CDIM)          // per-layer state slab, 786432
#define FD_EPS_F 1e-3f
#define LN_EPS_F 1e-5f

typedef __attribute__((ext_vector_type(16))) __bf16 v16bf;
typedef __attribute__((ext_vector_type(8)))  float  v8f;
typedef __attribute__((ext_vector_type(4)))  unsigned int u32x4;

union FragU { v16bf v; u32x4 u[2]; };

// LN modes (also map to gemm2 epilogue = mode+1)
#define MODE_WARM  0
#define MODE_PLAIN 1
#define MODE_PLUS  2
#define MODE_MINUS 3
// gemm epilogue modes
#define EPI_GELU_H 0   // gemm1: +b1, gelu, store bf16 h
#define EPI_WARM   1   // gemm2: +b2+x0, store states[l+1]
#define EPI_RES    2   // gemm2: +b2+x_l, res = y - states[l+1]
#define EPI_OP     3   // gemm2: +b2+(x_l+eps v), op = y
#define EPI_DIAG   4   // gemm2: +b2+(x_l-eps v), diagJ = (op-y)/(2eps)*v

__device__ __forceinline__ float rade(int it, uint32_t idx) {
  // deterministic Rademacher probe (stand-in for jax threefry)
  uint32_t h = idx ^ (0x9E3779B9u * (uint32_t)(it + 1));
  h *= 0x85EBCA6Bu; h ^= h >> 13; h *= 0xC2B2AE35u; h ^= h >> 16;
  return (h & 1u) ? 1.0f : -1.0f;
}

__device__ __forceinline__ float gelu_tanh(float x) {
  const float c0 = 0.7978845608028654f;     // sqrt(2/pi)
  float x3 = x * x * x;
  return 0.5f * x * (1.0f + tanhf(c0 * (x + 0.044715f * x3)));
}

// Async global->LDS copy of 32 contiguous bytes (two B128 ops; the immediate
// offset is added to BOTH the global and LDS addresses per the CDNA5 ISA).
__device__ __forceinline__ void async_copy32(unsigned lds_addr, const void* gptr) {
  unsigned long long ga = (unsigned long long)(uintptr_t)gptr;
  asm volatile("global_load_async_to_lds_b128 %0, %1, off\n\t"
               "global_load_async_to_lds_b128 %0, %1, off offset:16"
               :: "v"(lds_addr), "v"(ga) : "memory");
}

__device__ __forceinline__ void wait_async0() {
  asm volatile("s_wait_asynccnt 0x0" ::: "memory");
}

// ---------------------------------------------------------------------------
// Weight convert f32 -> bf16 with WMMA-B fragment swizzle.
// dst layout: [l][k/32][n/16][lane32][elem16], lane = ((k%32)/16)*16 + n%16,
// elem = k%16  -> B fragment for one 16x16x32 tile is 32 lanes x 16 contiguous
// bf16 (two 16B LDS loads per lane).
// ---------------------------------------------------------------------------
__global__ __launch_bounds__(256) void swizzle_weights(
    const float* __restrict__ src, __bf16* __restrict__ dst, int K, int N) {
  size_t idx = (size_t)blockIdx.x * 256 + threadIdx.x;
  size_t total = (size_t)L_LAYERS * K * N;
  if (idx >= total) return;
  int n = (int)(idx % N);
  size_t t = idx / N;
  int k = (int)(t % K);
  int l = (int)(t / K);
  float v = src[idx];
  size_t d = (size_t)l * K * N
           + (size_t)(k >> 5) * 32 * N
           + (size_t)(n >> 4) * 512
           + (size_t)((((k >> 4) & 1) << 4) + (n & 15)) * 16
           + (size_t)(k & 15);
  dst[d] = (__bf16)v;
}

// ---------------------------------------------------------------------------
// LayerNorm (+ optional +-eps*v perturbation) -> bf16. Grid (BT, L), 256 thr.
// ---------------------------------------------------------------------------
__global__ __launch_bounds__(256) void ln_kernel(
    const float* __restrict__ states, const float* __restrict__ g,
    const float* __restrict__ beta, __bf16* __restrict__ xn,
    int mode, int it) {
  __shared__ float s_sum[256];
  __shared__ float s_sq[256];
  int r = blockIdx.x, l = blockIdx.y, tidx = threadIdx.x;
  const float* xrow =
      states + (mode == MODE_WARM ? (size_t)0 : (size_t)l * S_ELEMS) + (size_t)r * CDIM;
  uint32_t pbase = (uint32_t)l * S_ELEMS + (uint32_t)r * CDIM;

  float xv[3], lsum = 0.f, lsq = 0.f;
#pragma unroll
  for (int j = 0; j < 3; ++j) {
    int c = tidx + j * 256;
    float x = xrow[c];
    if (mode == MODE_PLUS)       x += FD_EPS_F * rade(it, pbase + c);
    else if (mode == MODE_MINUS) x -= FD_EPS_F * rade(it, pbase + c);
    xv[j] = x; lsum += x; lsq += x * x;
  }
  s_sum[tidx] = lsum; s_sq[tidx] = lsq;
  __syncthreads();
  for (int off = 128; off > 0; off >>= 1) {
    if (tidx < off) { s_sum[tidx] += s_sum[tidx + off]; s_sq[tidx] += s_sq[tidx + off]; }
    __syncthreads();
  }
  float mu  = s_sum[0] * (1.0f / CDIM);
  float var = s_sq[0] * (1.0f / CDIM) - mu * mu;
  float rs  = rsqrtf(var + LN_EPS_F);
#pragma unroll
  for (int j = 0; j < 3; ++j) {
    int c = tidx + j * 256;
    float y = (xv[j] - mu) * rs * g[(size_t)l * CDIM + c] + beta[(size_t)l * CDIM + c];
    xn[(size_t)pbase + c] = (__bf16)y;
  }
}

// ---------------------------------------------------------------------------
// Batched bf16 WMMA GEMM: D = A(MxK, row-major bf16) * Bw(swizzled bf16).
// Workgroup: 128x128 tile, 8 waves (4 along M x 2 along N), each 32x64.
// Mainloop: BK=32, double-buffered LDS filled by async global->LDS B128
// (ASYNCcnt path), one barrier per stage, v_wmma_f32_16x16x32_bf16 core.
// ---------------------------------------------------------------------------
__global__ __launch_bounds__(256) void wmma_gemm(
    const __bf16* __restrict__ A, const __bf16* __restrict__ Bw,
    const float* __restrict__ bias, int M, int N, int K,
    int epimode, int it,
    const float* statesc, float* statesw,
    float* res, float* opw, const float* opr, float* diagJ,
    __bf16* hout) {
  __shared__ __bf16 As[2][128 * 32];   // row-major [r][k], 8KB each
  __shared__ __bf16 Bs[2][8 * 512];    // fragment-order, 8 n-tiles, 8KB each

  const int l     = blockIdx.z;
  const int mbase = blockIdx.x * 128;
  const int nbase = blockIdx.y * 128;
  const int tidx  = threadIdx.x;
  const int lane  = tidx & 31;
  const int wid   = tidx >> 5;
  const int waveM = wid & 3;    // 0..3 -> 32-row strip
  const int waveN = wid >> 2;   // 0..1 -> 64-col strip

  v8f acc[2][4];
#pragma unroll
  for (int mi = 0; mi < 2; ++mi)
#pragma unroll
    for (int ni = 0; ni < 4; ++ni)
      acc[mi][ni] = (v8f){0.f, 0.f, 0.f, 0.f, 0.f, 0.f, 0.f, 0.f};

  const __bf16* Ag = A  + (size_t)l * M * K;
  const __bf16* Bg = Bw + (size_t)l * K * N;

  const int aRow = tidx >> 1;        // 0..127
  const int aSeg = tidx & 1;         // which 16-bf16 half of the 32-wide K row
  const int m16  = lane & 15;
  const int half = lane >> 4;
  const int nstages = K >> 5;

  // per-thread LDS destinations (wave-relative addresses for the async ops)
  const unsigned ldsA0 = (unsigned)(uintptr_t)(&As[0][aRow * 32 + aSeg * 16]);
  const unsigned ldsA1 = (unsigned)(uintptr_t)(&As[1][aRow * 32 + aSeg * 16]);
  const unsigned ldsB0 = (unsigned)(uintptr_t)(&Bs[0][tidx * 16]);
  const unsigned ldsB1 = (unsigned)(uintptr_t)(&Bs[1][tidx * 16]);

  auto stage_async = [&](int buf, int ks) {
    const __bf16* agp = Ag + (size_t)(mbase + aRow) * K + ks * 32 + aSeg * 16;
    const __bf16* bgp = Bg + (size_t)ks * 32 * N + (size_t)(nbase >> 4) * 512 + tidx * 16;
    async_copy32(buf ? ldsA1 : ldsA0, agp);
    async_copy32(buf ? ldsB1 : ldsB0, bgp);
  };

  stage_async(0, 0);
  wait_async0();
  __syncthreads();

  for (int ks = 0; ks < nstages; ++ks) {
    const int cur = ks & 1;
    const bool more = (ks + 1 < nstages);
    if (more) stage_async(cur ^ 1, ks + 1);   // overlaps with WMMA below

    const __bf16* Asb = As[cur];
    const __bf16* Bsb = Bs[cur];
    FragU afr[2], bfr[4];
#pragma unroll
    for (int mi = 0; mi < 2; ++mi) {
      int rl = waveM * 32 + mi * 16 + m16;
      afr[mi].u[0] = *(const u32x4*)(Asb + rl * 32 + half * 16);       // K = half*16 .. +7
      afr[mi].u[1] = *(const u32x4*)(Asb + rl * 32 + half * 16 + 8);  // K = half*16+8 .. +15
    }
#pragma unroll
    for (int ni = 0; ni < 4; ++ni) {
      int nt = waveN * 4 + ni;
      bfr[ni].u[0] = *(const u32x4*)(Bsb + nt * 512 + lane * 16);
      bfr[ni].u[1] = *(const u32x4*)(Bsb + nt * 512 + lane * 16 + 8);
    }
#pragma unroll
    for (int mi = 0; mi < 2; ++mi)
#pragma unroll
      for (int ni = 0; ni < 4; ++ni)
        acc[mi][ni] = __builtin_amdgcn_wmma_f32_16x16x32_bf16(
            false, afr[mi].v, false, bfr[ni].v, (short)0, acc[mi][ni], false, false);

    if (more) wait_async0();   // next buffer's data landed in LDS
    __syncthreads();           // everyone done reading cur / sees nxt
  }

  // ----------------- epilogue -----------------
#pragma unroll
  for (int mi = 0; mi < 2; ++mi) {
#pragma unroll
    for (int ni = 0; ni < 4; ++ni) {
#pragma unroll
      for (int i = 0; i < 8; ++i) {
        int r = mbase + waveM * 32 + mi * 16 + i + half * 8;
        int n = nbase + waveN * 64 + ni * 16 + (lane & 15);
        float y = acc[mi][ni][i] + bias[(size_t)l * N + n];
        if (epimode == EPI_GELU_H) {
          hout[(size_t)l * M * N + (size_t)r * N + n] = (__bf16)gelu_tanh(y);
        } else {
          size_t e  = (size_t)r * N + n;          // N == CDIM here
          size_t pe = (size_t)l * S_ELEMS + e;
          float xin;
          if (epimode == EPI_WARM) {
            xin = statesc[e];                     // x0 slab for all layers
          } else {
            xin = statesc[pe];
            if (epimode == EPI_OP)        xin += FD_EPS_F * rade(it, (uint32_t)pe);
            else if (epimode == EPI_DIAG) xin -= FD_EPS_F * rade(it, (uint32_t)pe);
          }
          y += xin;
          if (epimode == EPI_WARM) {
            statesw[(size_t)(l + 1) * S_ELEMS + e] = y;
          } else if (epimode == EPI_RES) {
            res[pe] = y - statesc[(size_t)(l + 1) * S_ELEMS + e];
          } else if (epimode == EPI_OP) {
            opw[pe] = y;
          } else { // EPI_DIAG
            float v = rade(it, (uint32_t)pe);
            diagJ[pe] = (opr[pe] - y) * (0.5f / FD_EPS_F) * v;
          }
        }
      }
    }
  }
}

// ---------------------------------------------------------------------------
// Affine scan over L (x_l = A_l x_{l-1} + b_l, x_0 = b_0) + state update.
// ---------------------------------------------------------------------------
__global__ __launch_bounds__(256) void scan_update(
    const float* __restrict__ diagJ, const float* __restrict__ res,
    float* __restrict__ states) {
  int p = blockIdx.x * 256 + threadIdx.x;
  if (p >= S_ELEMS) return;
  float d = 0.f;
#pragma unroll
  for (int l = 0; l < L_LAYERS; ++l) {
    float b = res[(size_t)l * S_ELEMS + p];
    d = (l == 0) ? b : fmaf(diagJ[(size_t)l * S_ELEMS + p], d, b);
    states[(size_t)(l + 1) * S_ELEMS + p] += d;
  }
}

// ---------------------------------------------------------------------------
// Host orchestration
// ---------------------------------------------------------------------------
extern "C" void kernel_launch(void* const* d_in, const int* in_sizes, int n_in,
                              void* d_out, int out_size, void* d_ws, size_t ws_size,
                              hipStream_t stream) {
  (void)in_sizes; (void)n_in; (void)out_size; (void)ws_size;
  const float* x0   = (const float*)d_in[0];
  const float* w1   = (const float*)d_in[1];
  const float* b1   = (const float*)d_in[2];
  const float* w2   = (const float*)d_in[3];
  const float* b2   = (const float*)d_in[4];
  const float* g    = (const float*)d_in[5];
  const float* beta = (const float*)d_in[6];

  char* ws = (char*)d_ws;
  size_t off = 0;
  auto carve = [&](size_t bytes) -> char* {
    char* p = ws + off;
    off += (bytes + 255) & ~(size_t)255;
    return p;
  };
  const size_t WELEMS = (size_t)L_LAYERS * CDIM * FDIM;        // 28.3M
  __bf16* w1b   = (__bf16*)carve(WELEMS * 2);
  __bf16* w2b   = (__bf16*)carve(WELEMS * 2);
  float*  states= (float*) carve((size_t)(L_LAYERS + 1) * S_ELEMS * 4);
  __bf16* xn    = (__bf16*)carve((size_t)L_LAYERS * S_ELEMS * 2);
  __bf16* h     = (__bf16*)carve((size_t)L_LAYERS * BT * FDIM * 2);
  float*  res   = (float*) carve((size_t)L_LAYERS * S_ELEMS * 4);
  float*  op    = (float*) carve((size_t)L_LAYERS * S_ELEMS * 4);
  float*  diagJ = (float*) carve((size_t)L_LAYERS * S_ELEMS * 4);

  // 1. convert + swizzle weights to WMMA bf16 fragment order
  {
    int blocks = (int)((WELEMS + 255) / 256);
    swizzle_weights<<<blocks, 256, 0, stream>>>(w1, w1b, CDIM, FDIM);
    swizzle_weights<<<blocks, 256, 0, stream>>>(w2, w2b, FDIM, CDIM);
  }
  // 2. states[0] = x0
  hipMemcpyAsync(states, x0, (size_t)S_ELEMS * 4, hipMemcpyDeviceToDevice, stream);

  dim3 lnGrid(BT, L_LAYERS);
  dim3 g1Grid(BT / 128, FDIM / 128, L_LAYERS);   // (8,24,12)
  dim3 g2Grid(BT / 128, CDIM / 128, L_LAYERS);   // (8,6,12)

  auto run_blocks = [&](int lnmode, int it) {
    ln_kernel<<<lnGrid, 256, 0, stream>>>(states, g, beta, xn, lnmode, it);
    wmma_gemm<<<g1Grid, 256, 0, stream>>>(xn, w1b, b1, BT, FDIM, CDIM,
                                          EPI_GELU_H, it,
                                          nullptr, nullptr, nullptr, nullptr, nullptr, nullptr, h);
    wmma_gemm<<<g2Grid, 256, 0, stream>>>(h, w2b, b2, BT, CDIM, FDIM,
                                          lnmode + 1, it,
                                          states, states, res, op, op, diagJ, nullptr);
  };

  // 3. Picard warm start: states[l+1] = block_l(x0)
  run_blocks(MODE_WARM, 0);

  // 4. quasi-DEER iterations
  for (int it = 0; it < 2; ++it) {
    run_blocks(MODE_PLAIN, it);   // res = block(x_l) - x_{l+1}
    run_blocks(MODE_PLUS,  it);   // op  = block(x_l + eps v)
    run_blocks(MODE_MINUS, it);   // diagJ = (op - block(x_l - eps v))/(2eps) * v
    scan_update<<<S_ELEMS / 256, 256, 0, stream>>>(diagJ, res, states);
  }

  // 5. output = states[L]
  hipMemcpyAsync(d_out, states + (size_t)L_LAYERS * S_ELEMS,
                 (size_t)S_ELEMS * 4, hipMemcpyDeviceToDevice, stream);
}
